// FlexMoEModel_52166672777419
// MI455X (gfx1250) — matrix operand order repaired and checked by reading, not compiled, not run on hardware
//
#include <hip/hip_runtime.h>
#include <hip/hip_bf16.h>
#include <math.h>
#include <stdint.h>

typedef __attribute__((ext_vector_type(16))) __bf16    v16bf;
typedef __attribute__((ext_vector_type(8)))  float     v8f;
typedef __attribute__((ext_vector_type(4)))  unsigned  u32x4;

#define DEV __device__ __forceinline__

static constexpr int Bsz = 32, Ppm = 64, Dm = 768, Hh = 8, Ex = 8;
static constexpr int Sq  = 2 * Ppm;      // 128 tokens per batch element
static constexpr int Tt  = Bsz * Sq;     // 4096 tokens
static constexpr int Df  = 2 * Dm;       // 1536
static constexpr int Hd  = Dm / Hh;      // 96
static constexpr int Ldq = 3 * Dm;       // 2304 fused qkv row stride

union FragU { u32x4 u[2]; v16bf v; };

DEV v8f wmma_bf16(v16bf a, v16bf b, v8f c) {
  return __builtin_amdgcn_wmma_f32_16x16x32_bf16(false, a, false, b, (short)0, c,
                                                 false, false);
}

DEV float gelu_exact(float x) { return 0.5f * x * (1.0f + erff(x * 0.70710678118654752f)); }

// A fragment (16x32) from K-contiguous bf16 rows: lane holds row, K = hh*16 + {0..15}
// contiguous (lanes 0-15: 0..15, lanes 16-31: 16..31).
DEV v16bf frag_a_rows(const __bf16* __restrict__ base, int ld, int row, int k0, int hh) {
  const __bf16* p = base + (size_t)row * ld + k0 + hh * 16;
  FragU f;
  f.u[0] = *(const u32x4*)p;
  f.u[1] = *(const u32x4*)(p + 8);
  return f.v;
}

// B fragment (32x16) where B = W^T is stored K-contiguous per output column:
// lane holds col, K = hh*16 + {0..15} contiguous.
DEV v16bf frag_b_kcontig(const __bf16* __restrict__ base, int ld, int col, int k0, int hh) {
  const __bf16* p = base + (size_t)col * ld + k0 + hh * 16;
  FragU f;
  f.u[0] = *(const u32x4*)p;
  f.u[1] = *(const u32x4*)(p + 8);
  return f.v;
}

// B fragment from row-major [K][N] data (strided gather; only used for V in attention)
DEV v16bf frag_b_strided(const __bf16* __restrict__ base, int ld, int k0, int col, int hh) {
  const __bf16* p = base + (size_t)(k0 + hh * 16) * ld + col;
  v16bf b;
#pragma unroll
  for (int j = 0; j < 16; ++j) b[j] = p[(size_t)j * ld];
  return b;
}

DEV uint32_t lds_off(const void* p) { return (uint32_t)(uintptr_t)p; }

DEV void async_b128(uint32_t loff, const void* gp) {
  asm volatile("global_load_async_to_lds_b128 %0, %1, off"
               :: "v"(loff), "v"(gp) : "memory");
}
DEV void wait_async0() { asm volatile("s_wait_asynccnt 0x0" ::: "memory"); }

// ============ LDS-tiled bf16 WMMA GEMM, async double-buffered staging ============
// C[M,N] = epi( A[M,K] @ W[K,N] ), weights passed transposed: Bt[N][K] (ldb=K).
// ACT: 0 none / 1 gelu / 2 relu.  ACC: C += (f32 only).  BIAS: +bias[col].
// OUTBF: store bf16 instead of f32.
// GM: 0 dense; 1 gather A rows through idx[] (compact out, dynamic cnt);
//     2 A compact, scatter C rows through idx[] scaled by gval[] (dynamic cnt).
template <int ACT, bool ACC, bool BIAS, bool OUTBF, int GM>
__global__ void __launch_bounds__(256) gemm_tiled(
    const __bf16* __restrict__ A, int lda, const __bf16* __restrict__ Bt, int ldb,
    const float* __restrict__ bias, void* __restrict__ Cout, int ldc, int Mfix,
    const int* __restrict__ cntp, int N, int Kd, const int* __restrict__ idx,
    const float* __restrict__ gval, float outScale) {
  __shared__ __attribute__((aligned(16))) __bf16 Ash[2][128][32];
  __shared__ __attribute__((aligned(16))) __bf16 Bsh[2][128][32];

  const int tid = threadIdx.x;
  const int lane = tid & 31, wave = tid >> 5;
  const int lh = lane & 15, hh = lane >> 4;
  const int wm = wave >> 1, wn = wave & 1;        // 4x2 wave grid, 32x64 per wave
  const int m0 = blockIdx.y * 128, n0 = blockIdx.x * 128;
  const int cnt = (GM == 0) ? Mfix : *cntp;
  if (m0 >= cnt) return;  // block-uniform exit (before any barrier)

  auto stage = [&](int kb, int buf) {
    const int k0 = kb * 32;
    // A tile: 128 rows x 32 K  (512 x 16B chunks)
    for (int c = tid; c < 512; c += 256) {
      const int r = c >> 2, q = (c & 3) * 8;
      int row = m0 + r; if (row >= cnt) row = cnt - 1;
      const int arow = (GM == 1) ? idx[row] : row;
      async_b128(lds_off(&Ash[buf][r][q]), A + (size_t)arow * lda + k0 + q);
    }
    // Bt tile: 128 out-cols x 32 K
    for (int c = tid; c < 512; c += 256) {
      const int r = c >> 2, q = (c & 3) * 8;
      async_b128(lds_off(&Bsh[buf][r][q]), Bt + (size_t)(n0 + r) * ldb + k0 + q);
    }
  };

  v8f acc[2][4];
#pragma unroll
  for (int s = 0; s < 2; ++s)
#pragma unroll
    for (int t = 0; t < 4; ++t) acc[s][t] = {};

  const int nk = Kd / 32;
  stage(0, 0);
  for (int kb = 0; kb < nk; ++kb) {
    const int buf = kb & 1;
    wait_async0();
    __syncthreads();                       // tile[kb] visible to all; prev reads done
    if (kb + 1 < nk) stage(kb + 1, buf ^ 1);  // overlap next copy with this compute

    FragU afr[2], bfr[4];
#pragma unroll
    for (int s = 0; s < 2; ++s) {
      const int row = wm * 32 + s * 16 + lh;
      afr[s].u[0] = *(const u32x4*)&Ash[buf][row][hh * 16];
      afr[s].u[1] = *(const u32x4*)&Ash[buf][row][hh * 16 + 8];
    }
#pragma unroll
    for (int t = 0; t < 4; ++t) {
      const int col = wn * 64 + t * 16 + lh;
      bfr[t].u[0] = *(const u32x4*)&Bsh[buf][col][hh * 16];
      bfr[t].u[1] = *(const u32x4*)&Bsh[buf][col][hh * 16 + 8];
    }
#pragma unroll
    for (int s = 0; s < 2; ++s)
#pragma unroll
      for (int t = 0; t < 4; ++t)
        acc[s][t] = wmma_bf16(afr[s].v, bfr[t].v, acc[s][t]);
  }

#pragma unroll
  for (int s = 0; s < 2; ++s) {
    const int rowB = m0 + wm * 32 + s * 16 + hh * 8;
#pragma unroll
    for (int t = 0; t < 4; ++t) {
      const int col = n0 + wn * 64 + t * 16 + lh;
      const float bb = BIAS ? bias[col] : 0.0f;
#pragma unroll
      for (int r = 0; r < 8; ++r) {
        const int rr = rowB + r;
        if (rr < cnt) {
          float v = acc[s][t][r] + bb;
          if (ACT == 1) v = gelu_exact(v);
          if (ACT == 2) v = fmaxf(v, 0.0f);
          v *= outScale;
          int crow = rr;
          if (GM == 2) { v *= gval[rr]; crow = idx[rr]; }
          if (OUTBF) {
            ((__bf16*)Cout)[(size_t)crow * ldc + col] = (__bf16)v;
          } else {
            float* cp = (float*)Cout + (size_t)crow * ldc + col;
            if (ACC) *cp += v; else *cp = v;
          }
        }
      }
    }
  }
}

template <int ACT, bool ACC, bool BIAS, bool OUTBF, int GM>
static inline void lg(const __bf16* A, int lda, const __bf16* Bt, int ldb,
                      const float* bias, void* C, int ldc, int M, const int* cntp,
                      int N, int K, const int* idx, const float* gv, float osc,
                      hipStream_t st) {
  dim3 g(N / 128, (M + 127) / 128);
  gemm_tiled<ACT, ACC, BIAS, OUTBF, GM><<<g, 256, 0, st>>>(
      A, lda, Bt, ldb, bias, C, ldc, M, cntp, N, K, idx, gv, osc);
}

// ============ fused attention: scores WMMA -> softmax -> P@V WMMA (all bf16 in) ============
__global__ void __launch_bounds__(256) attn_kernel(const __bf16* __restrict__ qkv,
                                                   __bf16* __restrict__ out) {
  __shared__ __attribute__((aligned(16))) __bf16 Pl[8][16][128];  // 32 KB
  const int bh = blockIdx.x, b = bh >> 3, h = bh & 7;
  const int lane = threadIdx.x & 31, wave = threadIdx.x >> 5;
  const int lh = lane & 15, hh = lane >> 4;
  const __bf16* base = qkv + (size_t)b * Sq * Ldq;
  const __bf16* qp = base + h * Hd;
  const __bf16* kp = base + Dm + h * Hd;
  const __bf16* vp = base + 2 * Dm + h * Hd;
  const int m0 = wave * 16;
  const float scale = 0.10206207261596577f;  // 1/sqrt(96)

  v8f accS[8];
#pragma unroll
  for (int ct = 0; ct < 8; ++ct) {
    v8f acc = {};
#pragma unroll
    for (int kc = 0; kc < 3; ++kc) {
      const int k0 = kc * 32;
      v16bf a = frag_a_rows(qp, Ldq, m0 + lh, k0, hh);
      v16bf bb = frag_b_kcontig(kp, Ldq, ct * 16 + lh, k0, hh);  // B = K^T
      acc = wmma_bf16(a, bb, acc);
    }
#pragma unroll
    for (int r = 0; r < 8; ++r) acc[r] *= scale;
    accS[ct] = acc;
  }

  float mx[8], sm[8];
#pragma unroll
  for (int r = 0; r < 8; ++r) {
    float m = accS[0][r];
#pragma unroll
    for (int ct = 1; ct < 8; ++ct) m = fmaxf(m, accS[ct][r]);
#pragma unroll
    for (int msk = 1; msk < 16; msk <<= 1) m = fmaxf(m, __shfl_xor(m, msk, 32));
    mx[r] = m;
  }
#pragma unroll
  for (int r = 0; r < 8; ++r) {
    float s = 0.0f;
#pragma unroll
    for (int ct = 0; ct < 8; ++ct) {
      float e = expf(accS[ct][r] - mx[r]);
      accS[ct][r] = e;
      s += e;
    }
#pragma unroll
    for (int msk = 1; msk < 16; msk <<= 1) s += __shfl_xor(s, msk, 32);
    sm[r] = 1.0f / s;
  }
#pragma unroll
  for (int ct = 0; ct < 8; ++ct)
#pragma unroll
    for (int r = 0; r < 8; ++r)
      Pl[wave][hh * 8 + r][ct * 16 + lh] = (__bf16)(accS[ct][r] * sm[r]);
  __syncthreads();

#pragma unroll
  for (int nt = 0; nt < 6; ++nt) {
    v8f acc = {};
#pragma unroll
    for (int kc = 0; kc < 4; ++kc) {
      const int k0 = kc * 32;
      FragU a;
      a.u[0] = *(const u32x4*)&Pl[wave][lh][k0 + hh * 16];
      a.u[1] = *(const u32x4*)&Pl[wave][lh][k0 + hh * 16 + 8];
      v16bf bb = frag_b_strided(vp, Ldq, k0, nt * 16 + lh, hh);
      acc = wmma_bf16(a.v, bb, acc);
    }
#pragma unroll
    for (int r = 0; r < 8; ++r)
      out[(size_t)(b * Sq + m0 + hh * 8 + r) * Dm + h * Hd + nt * 16 + lh] =
          (__bf16)acc[r];
  }
}

// ============ elementwise / reduction kernels ============

__global__ void tconv_kernel(const float* __restrict__ W, __bf16* __restrict__ Wt,
                             int K, int N) {
  int i = blockIdx.x * blockDim.x + threadIdx.x;
  if (i >= K * N) return;
  int k = i / N, n = i % N;
  Wt[(size_t)n * K + k] = (__bf16)W[i];  // store transposed: [N][K]
}

__global__ void prep_kernel(const float* __restrict__ x0, const float* __restrict__ x1,
                            const float* __restrict__ pos, float* __restrict__ xcat) {
  int i = blockIdx.x * blockDim.x + threadIdx.x;
  if (i >= Tt * Dm) return;
  int d = i % Dm, row = i / Dm, b = row >> 7, sI = row & 127;
  float p = pos[sI * Dm + d];
  float v = (sI < 64) ? x0[(b * 64 + sI) * Dm + d] : x1[(b * 64 + (sI - 64)) * Dm + d];
  xcat[i] = v + p;
}

__global__ void ln_kernel(const float* __restrict__ x, const float* __restrict__ w,
                          const float* __restrict__ b, __bf16* __restrict__ y) {
  const int row = blockIdx.x;
  const float* xr = x + (size_t)row * Dm;
  float s = 0.f, s2 = 0.f;
  for (int i = threadIdx.x; i < Dm; i += blockDim.x) {
    float v = xr[i];
    s += v; s2 += v * v;
  }
  __shared__ float rs[256], rq[256];
  rs[threadIdx.x] = s; rq[threadIdx.x] = s2;
  __syncthreads();
  for (int off = 128; off > 0; off >>= 1) {
    if ((int)threadIdx.x < off) {
      rs[threadIdx.x] += rs[threadIdx.x + off];
      rq[threadIdx.x] += rq[threadIdx.x + off];
    }
    __syncthreads();
  }
  const float mu = rs[0] / Dm;
  const float inv = rsqrtf(rq[0] / Dm - mu * mu + 1e-5f);
  for (int i = threadIdx.x; i < Dm; i += blockDim.x)
    y[(size_t)row * Dm + i] = (__bf16)((xr[i] - mu) * inv * w[i] + b[i]);
}

__global__ void zero_kernel(int* __restrict__ cnt) {
  if (threadIdx.x < Ex) cnt[threadIdx.x] = 0;
}

// top-2 gating -> per-expert compact token lists (order-independent result)
__global__ void gate_kernel(const __bf16* __restrict__ x, const float* __restrict__ wg,
                            int* __restrict__ cnt, int* __restrict__ idx,
                            float* __restrict__ gval) {
  int t = blockIdx.x * blockDim.x + threadIdx.x;
  if (t >= Tt) return;
  float lgt[Ex];
#pragma unroll
  for (int e = 0; e < Ex; ++e) lgt[e] = 0.0f;
  const __bf16* xr = x + (size_t)t * Dm;
  for (int d = 0; d < Dm; ++d) {
    float xv = (float)xr[d];
#pragma unroll
    for (int e = 0; e < Ex; ++e) lgt[e] += xv * wg[d * Ex + e];
  }
  int i1 = 0;
#pragma unroll
  for (int e = 1; e < Ex; ++e) if (lgt[e] > lgt[i1]) i1 = e;
  int i2 = (i1 == 0) ? 1 : 0;
#pragma unroll
  for (int e = 0; e < Ex; ++e) if (e != i1 && lgt[e] > lgt[i2]) i2 = e;
  float e2 = expf(lgt[i2] - lgt[i1]);
  float inv = 1.0f / (1.0f + e2);
  int p1 = atomicAdd(&cnt[i1], 1);
  idx[i1 * Tt + p1] = t; gval[i1 * Tt + p1] = inv;
  int p2 = atomicAdd(&cnt[i2], 1);
  idx[i2 * Tt + p2] = t; gval[i2 * Tt + p2] = e2 * inv;
}

__global__ void pool_kernel(const float* __restrict__ x, __bf16* __restrict__ pooled) {
  int i = blockIdx.x * blockDim.x + threadIdx.x;
  if (i >= Bsz * Df) return;
  int b = i / Df, c = i % Df, half = c / Dm, d = c % Dm;
  const float* base = x + (size_t)(b * Sq + half * Ppm) * Dm + d;
  float s = 0.0f;
  for (int t = 0; t < Ppm; ++t) s += base[(size_t)t * Dm];
  pooled[i] = (__bf16)(s * (1.0f / (float)Ppm));
}

__global__ void head2_kernel(const float* __restrict__ hh_, const float* __restrict__ w2,
                             const float* __restrict__ b2, float* __restrict__ out) {
  int i = threadIdx.x;
  if (i >= Bsz * 2) return;
  int b = i >> 1, j = i & 1;
  float s = b2[j];
  for (int d = 0; d < Dm; ++d) s += hh_[(size_t)b * Dm + d] * w2[d * 2 + j];
  out[i] = s;
}

// ============ host orchestration ============

static inline void tconv(const float* W, __bf16* Wt, int K, int N, hipStream_t st) {
  int n = K * N;
  tconv_kernel<<<(n + 255) / 256, 256, 0, st>>>(W, Wt, K, N);
}

extern "C" void kernel_launch(void* const* d_in, const int* in_sizes, int n_in,
                              void* d_out, int out_size, void* d_ws, size_t ws_size,
                              hipStream_t stream) {
  (void)in_sizes; (void)n_in; (void)out_size; (void)ws_size;
  const float* x0      = (const float*)d_in[0];
  const float* x1      = (const float*)d_in[1];
  const float* pos     = (const float*)d_in[2];
  const float* ln1_w   = (const float*)d_in[3];
  const float* ln1_b   = (const float*)d_in[4];
  const float* ln2_w   = (const float*)d_in[5];
  const float* ln2_b   = (const float*)d_in[6];
  const float* wq      = (const float*)d_in[7];
  const float* wkv     = (const float*)d_in[8];
  const float* wo      = (const float*)d_in[9];
  const float* bo      = (const float*)d_in[10];
  const float* moe_wg  = (const float*)d_in[11];
  const float* moe_w1  = (const float*)d_in[12];
  const float* moe_b1  = (const float*)d_in[13];
  const float* moe_w2  = (const float*)d_in[14];
  const float* moe_b2  = (const float*)d_in[15];
  const float* mlp_w1  = (const float*)d_in[16];
  const float* mlp_b1  = (const float*)d_in[17];
  const float* mlp_w2  = (const float*)d_in[18];
  const float* mlp_b2  = (const float*)d_in[19];
  const float* head_w1 = (const float*)d_in[20];
  const float* head_b1 = (const float*)d_in[21];
  const float* head_w2 = (const float*)d_in[22];
  const float* head_b2 = (const float*)d_in[23];

  char* p = (char*)d_ws;
  auto take = [&](size_t bytes) -> void* {
    void* r = (void*)p;
    p += (bytes + 255) & ~(size_t)255;
    return r;
  };
  float*   xcat   = (float*)take((size_t)Tt * Dm * 4);
  __bf16*  lnout  = (__bf16*)take((size_t)Tt * Dm * 2);
  __bf16*  s1     = (__bf16*)take((size_t)Tt * Ldq * 2);  // qkv, reused as FFN hidden
  __bf16*  s2     = (__bf16*)take((size_t)Tt * Dm * 2);
  int*     idxB   = (int*)take((size_t)Ex * Tt * 4);
  float*   gvalB  = (float*)take((size_t)Ex * Tt * 4);
  int*     cnt    = (int*)take(256);
  __bf16*  pooled = (__bf16*)take((size_t)Bsz * Df * 2);
  float*   headh  = (float*)take((size_t)Bsz * Dm * 4);
  __bf16*  wqT    = (__bf16*)take((size_t)4 * Dm * Dm * 2);
  __bf16*  wkvT   = (__bf16*)take((size_t)4 * Dm * Df * 2);
  __bf16*  woT    = (__bf16*)take((size_t)4 * Dm * Dm * 2);
  __bf16*  m1T    = (__bf16*)take((size_t)16 * Dm * Df * 2);
  __bf16*  m2T    = (__bf16*)take((size_t)16 * Df * Dm * 2);
  __bf16*  p1T    = (__bf16*)take((size_t)2 * Dm * Df * 2);
  __bf16*  p2T    = (__bf16*)take((size_t)2 * Df * Dm * 2);
  __bf16*  h1T    = (__bf16*)take((size_t)Df * Dm * 2);

  // one-time weight transpose+convert to bf16 (Wt[N][K])
  for (int l = 0; l < 4; ++l) {
    tconv(wq + (size_t)l * Dm * Dm, wqT + (size_t)l * Dm * Dm, Dm, Dm, stream);
    tconv(wkv + (size_t)l * Dm * Df, wkvT + (size_t)l * Dm * Df, Dm, Df, stream);
    tconv(wo + (size_t)l * Dm * Dm, woT + (size_t)l * Dm * Dm, Dm, Dm, stream);
  }
  for (int w = 0; w < 16; ++w) {
    tconv(moe_w1 + (size_t)w * Dm * Df, m1T + (size_t)w * Dm * Df, Dm, Df, stream);
    tconv(moe_w2 + (size_t)w * Df * Dm, m2T + (size_t)w * Df * Dm, Df, Dm, stream);
  }
  for (int w = 0; w < 2; ++w) {
    tconv(mlp_w1 + (size_t)w * Dm * Df, p1T + (size_t)w * Dm * Df, Dm, Df, stream);
    tconv(mlp_w2 + (size_t)w * Df * Dm, p2T + (size_t)w * Df * Dm, Df, Dm, stream);
  }
  tconv(head_w1, h1T, Df, Dm, stream);

  prep_kernel<<<(Tt * Dm + 255) / 256, 256, 0, stream>>>(x0, x1, pos, xcat);

  for (int l = 0; l < 4; ++l) {
    ln_kernel<<<Tt, 256, 0, stream>>>(xcat, ln1_w + l * Dm, ln1_b + l * Dm, lnout);
    // fused QKV (bf16 out): q -> s1[:,0:768], kv -> s1[:,768:2304]
    lg<0, false, false, true, 0>(lnout, Dm, wqT + (size_t)l * Dm * Dm, Dm, nullptr,
                                 s1, Ldq, Tt, nullptr, Dm, Dm, nullptr, nullptr,
                                 1.0f, stream);
    lg<0, false, false, true, 0>(lnout, Dm, wkvT + (size_t)l * Dm * Df, Dm, nullptr,
                                 s1 + Dm, Ldq, Tt, nullptr, Df, Dm, nullptr, nullptr,
                                 1.0f, stream);
    attn_kernel<<<Bsz * Hh, 256, 0, stream>>>(s1, s2);
    // xcat = 2*(attn @ wo + bo)
    lg<0, false, true, false, 0>(s2, Dm, woT + (size_t)l * Dm * Dm, Dm, bo + l * Dm,
                                 xcat, Dm, Tt, nullptr, Dm, Dm, nullptr, nullptr,
                                 2.0f, stream);
    ln_kernel<<<Tt, 256, 0, stream>>>(xcat, ln2_w + l * Dm, ln2_b + l * Dm, lnout);
    if ((l & 1) == 0) {  // sparse MoE: gather / scatter by expert token lists
      const int sidx = l / 2;
      zero_kernel<<<1, 32, 0, stream>>>(cnt);
      gate_kernel<<<(Tt + 255) / 256, 256, 0, stream>>>(
          lnout, moe_wg + (size_t)sidx * Dm * Ex, cnt, idxB, gvalB);
      for (int e = 0; e < Ex; ++e) {
        const size_t wi = (size_t)(sidx * Ex + e);
        lg<1, false, true, true, 1>(lnout, Dm, m1T + wi * Dm * Df, Dm,
                                    moe_b1 + wi * Df, s1, Df, Tt, cnt + e, Df, Dm,
                                    idxB + e * Tt, nullptr, 1.0f, stream);
        lg<0, true, true, false, 2>(s1, Df, m2T + wi * Df * Dm, Df, moe_b2 + wi * Dm,
                                    xcat, Dm, Tt, cnt + e, Dm, Df, idxB + e * Tt,
                                    gvalB + e * Tt, 1.0f, stream);
      }
    } else {  // dense MLP
      const int m = l / 2;
      lg<1, false, true, true, 0>(lnout, Dm, p1T + (size_t)m * Dm * Df, Dm,
                                  mlp_b1 + m * Df, s1, Df, Tt, nullptr, Df, Dm,
                                  nullptr, nullptr, 1.0f, stream);
      lg<0, true, true, false, 0>(s1, Df, p2T + (size_t)m * Df * Dm, Df,
                                  mlp_b2 + m * Dm, xcat, Dm, Tt, nullptr, Dm, Df,
                                  nullptr, nullptr, 1.0f, stream);
    }
  }

  pool_kernel<<<(Bsz * Df + 255) / 256, 256, 0, stream>>>(xcat, pooled);
  lg<2, false, true, false, 0>(pooled, Df, h1T, Df, head_b1, headh, Dm, Bsz, nullptr,
                               Dm, Df, nullptr, nullptr, 1.0f, stream);
  head2_kernel<<<1, 64, 0, stream>>>(headh, head_w2, head_b2, (float*)d_out);
}
